// Depth_Conv2d_23218593202764
// MI455X (gfx1250) — matrix operand order repaired and checked by reading, not compiled, not run on hardware
//
#include <hip/hip_runtime.h>

typedef __attribute__((ext_vector_type(16))) _Float16 v16h;
typedef __attribute__((ext_vector_type(8)))  _Float16 v8h;
typedef __attribute__((ext_vector_type(8)))  float    v8f;

#define C_CH   384
#define HW     3136            // 56*56
#define NPIX   (32 * HW)       // 100352
#define QMAX   7.0f
#define SCALE_MIN 2e-16f

// ---------------- fake-quant: depthwise weights (per-channel over 9 taps) ----------------
__global__ void quant_dw_kernel(const float* __restrict__ W, float* __restrict__ Dq) {
  int c = blockIdx.x * blockDim.x + threadIdx.x;
  if (c >= C_CH) return;
  const float* w = W + c * 9;
  float m = 0.f;
#pragma unroll
  for (int i = 0; i < 9; ++i) m = fmaxf(m, fabsf(w[i]));
  float scale = fmaxf(m / QMAX, SCALE_MIN);
#pragma unroll
  for (int i = 0; i < 9; ++i) {
    float q = rintf(w[i] / scale);              // round-half-even, matches jnp.round
    q = fminf(fmaxf(q, -QMAX), QMAX);
    Dq[c * 9 + i] = q * scale;
  }
}

// ---------------- fake-quant: pointwise weights (per out-channel over 384) -> f16 A matrix ----------------
__global__ void quant_pw_kernel(const float* __restrict__ W, _Float16* __restrict__ Aq) {
  __shared__ float red[128];
  int co = blockIdx.x;
  const float* w = W + co * C_CH;
  float m = 0.f;
  for (int i = threadIdx.x; i < C_CH; i += 128) m = fmaxf(m, fabsf(w[i]));
  red[threadIdx.x] = m;
  __syncthreads();
  for (int s = 64; s > 0; s >>= 1) {
    if (threadIdx.x < (unsigned)s) red[threadIdx.x] = fmaxf(red[threadIdx.x], red[threadIdx.x + s]);
    __syncthreads();
  }
  float scale = fmaxf(red[0] / QMAX, SCALE_MIN);
  for (int i = threadIdx.x; i < C_CH; i += 128) {
    float q = rintf(w[i] / scale);
    q = fminf(fmaxf(q, -QMAX), QMAX);
    Aq[co * C_CH + i] = (_Float16)(q * scale);  // int4*scale fits f16 exactly (scale rounding only)
  }
}

// ---------------- depthwise 3x3 pad1 + NCHW -> NHWC(f16) transpose ----------------
// block = 256 threads, tile = 16 channels x 8h x 8w; LDS halo tile 16x10x10 (padded to 13).
// x is streamed exactly once -> non-temporal loads so the 154MB stream doesn't evict Y from L2.
__global__ __launch_bounds__(256) void dwconv_kernel(const float* __restrict__ X,
                                                     const float* __restrict__ Dq,
                                                     _Float16* __restrict__ Y) {
  __shared__ float tile[16][10][13];
  const int n  = blockIdx.z / 24;
  const int c0 = (blockIdx.z % 24) * 16;
  const int h0 = blockIdx.y * 8;
  const int w0 = blockIdx.x * 8;
  const int tid = threadIdx.x;

  for (int i = tid; i < 1600; i += 256) {       // 16 * 10 * 10 halo elements
    int cc = i / 100, r = i % 100;
    int hh = r / 10, ww = r % 10;
    int h = h0 + hh - 1, w = w0 + ww - 1;
    float v = 0.f;
    if (h >= 0 && h < 56 && w >= 0 && w < 56)
      v = __builtin_nontemporal_load(&X[(((size_t)n * C_CH + c0 + cc) * 56 + h) * 56 + w]);
    tile[cc][hh][ww] = v;
  }
  __syncthreads();

  const int cc = tid & 15;        // channel fastest -> contiguous f16 NHWC stores
  const int t  = tid >> 4;        // 0..15
  const int wq = t & 7;
  const int h2 = t >> 3;          // 0 or 1; each thread does h = h2, h2+2, h2+4, h2+6

  float wgt[9];
#pragma unroll
  for (int j = 0; j < 9; ++j) wgt[j] = Dq[(c0 + cc) * 9 + j];

  for (int hh = h2; hh < 8; hh += 2) {
    float s = 0.f;
#pragma unroll
    for (int dh = 0; dh < 3; ++dh)
#pragma unroll
      for (int dw = 0; dw < 3; ++dw)
        s += tile[cc][hh + dh][wq + dw] * wgt[dh * 3 + dw];
    size_t pix = (size_t)n * HW + (size_t)(h0 + hh) * 56 + (w0 + wq);
    Y[pix * C_CH + c0 + cc] = (_Float16)s;      // regular TH: keep Y resident in 192MB L2
  }
}

// ---------------- pointwise 1x1 as WMMA GEMM: Z[co, pix] = A[co, ci] * Y[pix, ci]^T + bias ----------------
// 8 waves/block; wave = two 16-pixel tiles x 64 co (8 wmma accumulators); K = 384 in steps of 32.
// A fragments are reused across both pixel tiles -> 12 b128 loads per 8 WMMA.
__global__ __launch_bounds__(256) void pw_gemm_kernel(const _Float16* __restrict__ A,
                                                      const _Float16* __restrict__ Y,
                                                      const float* __restrict__ bias,
                                                      float* __restrict__ Z) {
  const int lane = threadIdx.x & 31;
  const int wave = threadIdx.x >> 5;
  const int l16  = lane & 15;
  const int half = lane >> 4;                      // which 16-lane half of the wave
  const int pixBase = blockIdx.x * 256 + wave * 32;
  const int coBase  = blockIdx.y * 64;

  v8f acc[4][2] = {};                              // [co sub-tile][pixel tile]

  // B fragment source: Y is [pixel][ci] so K is contiguous per lane.
  // B layout used here (16-bit, 32x16, contiguous K, same as A): lane n = lane&15; K = half*16 + e  -> one 32B load.
  const _Float16* yrow0 = Y + (size_t)(pixBase + l16) * C_CH + half * 16;
  const _Float16* yrow1 = yrow0 + (size_t)16 * C_CH;
  // A layout (16-bit, 16x32): lane m = lane&15; kbase = half*16;
  //   elements 0..15 -> K = kbase+e (matches B)  -> two 16B loads.
  const int kbase = half * 16;

  for (int kc = 0; kc < C_CH; kc += 32) {
    v16h b0 = *(const v16h*)(yrow0 + kc);
    v16h b1 = *(const v16h*)(yrow1 + kc);
    union { v16h v; v8h h[2]; } af[4];
#pragma unroll
    for (int s = 0; s < 4; ++s) {
      const _Float16* arow = A + (size_t)(coBase + s * 16 + l16) * C_CH + kc + kbase;
      af[s].h[0] = *(const v8h*)(arow);
      af[s].h[1] = *(const v8h*)(arow + 8);
    }
#pragma unroll
    for (int s = 0; s < 4; ++s) {
      acc[s][0] = __builtin_amdgcn_wmma_f32_16x16x32_f16(false, af[s].v, false, b0,
                                                         (short)0, acc[s][0], false, false);
      acc[s][1] = __builtin_amdgcn_wmma_f32_16x16x32_f16(false, af[s].v, false, b1,
                                                         (short)0, acc[s][1], false, false);
    }
  }

  // D layout: VGPR r, lane -> M = r + half*8, N = l16. 3136 % 16 == 0 so each
  // 16-pixel tile sits inside one image and one contiguous z row per co.
  // z is write-once streamed -> non-temporal stores (don't evict Y / A from L2).
#pragma unroll
  for (int p = 0; p < 2; ++p) {
    const int pb = pixBase + p * 16;
    const int n  = pb / HW;
    const int hw = pb - n * HW + l16;
#pragma unroll
    for (int s = 0; s < 4; ++s) {
#pragma unroll
      for (int r = 0; r < 8; ++r) {
        int co = coBase + s * 16 + r + half * 8;
        __builtin_nontemporal_store(acc[s][p][r] + bias[co],
                                    &Z[((size_t)n * C_CH + co) * HW + hw]);
      }
    }
  }
}

extern "C" void kernel_launch(void* const* d_in, const int* in_sizes, int n_in,
                              void* d_out, int out_size, void* d_ws, size_t ws_size,
                              hipStream_t stream) {
  (void)in_sizes; (void)n_in; (void)out_size; (void)ws_size;
  const float* x    = (const float*)d_in[0];
  const float* dw_w = (const float*)d_in[1];
  const float* pw_w = (const float*)d_in[2];
  const float* pw_b = (const float*)d_in[3];
  float* z = (float*)d_out;

  // workspace layout (all offsets 32B-aligned for v16h loads)
  char* ws = (char*)d_ws;
  _Float16* Aq = (_Float16*)ws;                       // 384*384*2   = 294912 B
  float*    Dq = (float*)(ws + 294912);               // 384*9*4     = 13824 B
  _Float16* Yh = (_Float16*)(ws + 294912 + 13824);    // 100352*384*2 ≈ 73.5 MB (fits in 192MB L2)

  quant_dw_kernel<<<6, 64, 0, stream>>>(dw_w, Dq);
  quant_pw_kernel<<<C_CH, 128, 0, stream>>>(pw_w, Aq);
  dwconv_kernel<<<dim3(7, 7, 32 * 24), 256, 0, stream>>>(x, Dq, Yh);
  pw_gemm_kernel<<<dim3(NPIX / 256, C_CH / 64), 256, 0, stream>>>(Aq, Yh, pw_b, z);
}
